// AL_Attention_Block_37331855737373
// MI455X (gfx1250) — matrix operand order repaired and checked by reading, not compiled, not run on hardware
//
#include <hip/hip_runtime.h>
#include <hip/hip_bf16.h>
#include <stdint.h>

typedef __bf16 bf16;
typedef __bf16 bf16x16 __attribute__((ext_vector_type(16)));
typedef __bf16 bf16x8  __attribute__((ext_vector_type(8)));
typedef float  f32x8   __attribute__((ext_vector_type(8)));
typedef float  f32x4   __attribute__((ext_vector_type(4)));

// Problem constants
constexpr int CB = 32, CT = 50, CL = 128, CA = 64, CH = 512, CHEADS = 8, CDQ = 64, CDFF = 1024;
constexpr int MQ = CB * CT * CL;   // 204800 rows for Q-side GEMMs
constexpr int MK = CB * CT * CA;   // 102400 rows for K/V-side GEMMs

union FragB16 { bf16x16 v; bf16x8 h2[2]; };

__device__ inline bf16x8 cvt8p(const float* __restrict__ p) {
  f32x4 a = *(const f32x4*)p;
  f32x4 b = *(const f32x4*)(p + 4);
  bf16x8 r;
  r[0] = (bf16)a.x; r[1] = (bf16)a.y; r[2] = (bf16)a.z; r[3] = (bf16)a.w;
  r[4] = (bf16)b.x; r[5] = (bf16)b.y; r[6] = (bf16)b.z; r[7] = (bf16)b.w;
  return r;
}

__device__ inline bf16x8 cvt8v(f32x4 a, f32x4 b) {
  bf16x8 r;
  r[0] = (bf16)a.x; r[1] = (bf16)a.y; r[2] = (bf16)a.z; r[3] = (bf16)a.w;
  r[4] = (bf16)b.x; r[5] = (bf16)b.y; r[6] = (bf16)b.z; r[7] = (bf16)b.w;
  return r;
}

enum { EPI_QHEAD = 0, EPI_KHEAD = 1, EPI_VT = 2, EPI_BIAS_RES = 3, EPI_BIAS_RELU = 4 };

// C[M,N] = A[M,K](f32) @ W[N,K]^T (f32) + bias, with epilogue variants.
// Block tile 128x128, 8 waves (256 threads), K-step 32, double-buffered LDS,
// bf16 WMMA with f32 accumulation. Grid: (N/128, M/128) so that column blocks
// sharing the same A row-strip are schedule-adjacent (A served from L2).
template<int EPI, int SEQ>
__global__ __launch_bounds__(256)
void gemm_wmma(const float* __restrict__ Amat, const float* __restrict__ Wmat,
               const float* __restrict__ bias, const float* __restrict__ res,
               void* __restrict__ out, int N, int K)
{
  __shared__ __align__(16) bf16 sA[2][128 * 32];
  __shared__ __align__(16) bf16 sB[2][128 * 32];

  const int tid   = threadIdx.x;
  const int lane  = tid & 31;
  const int wave  = tid >> 5;
  const int l15   = lane & 15;
  const int khalf = lane >> 4;

  const int colBase = blockIdx.x * 128;
  const int rowBase = blockIdx.y * 128;
  const int wm = (wave & 3) * 32;   // wave's row offset inside block tile (2 M-tiles)
  const int wn = (wave >> 2) * 64;  // wave's col offset inside block tile (4 N-tiles)

  // Staging map: thread covers 16 consecutive floats of a 128x32 tile.
  const int ar = tid >> 1;
  const int ac = (tid & 1) * 16;
  const float* gA = Amat + (int64_t)(rowBase + ar) * K + ac;
  const float* gW = Wmat + (int64_t)(colBase + ar) * K + ac;
  bf16* lA = (bf16*)sA + ar * 32 + ac;   // + buf*128*32
  bf16* lW = (bf16*)sB + ar * 32 + ac;

  // Prologue: stage K-tile 0 into buffer 0.
  {
    *(bf16x8*)(lA)     = cvt8p(gA);
    *(bf16x8*)(lA + 8) = cvt8p(gA + 8);
    *(bf16x8*)(lW)     = cvt8p(gW);
    *(bf16x8*)(lW + 8) = cvt8p(gW + 8);
  }
  __syncthreads();

  f32x8 acc[2][4];
#pragma unroll
  for (int i = 0; i < 2; i++)
#pragma unroll
    for (int j = 0; j < 4; j++)
      acc[i][j] = (f32x8){0.f, 0.f, 0.f, 0.f, 0.f, 0.f, 0.f, 0.f};

  const int nk = K >> 5;
  for (int kt = 0; kt < nk; ++kt) {
    const int cur = kt & 1;
    const int nxt = cur ^ 1;
    const bool havenext = (kt + 1) < nk;

    // Issue next tile's global loads early (overlap with WMMA below).
    f32x4 pa0, pa1, pa2, pa3, pw0, pw1, pw2, pw3;
    if (havenext) {
      const float* ga = gA + (kt + 1) * 32;
      const float* gw = gW + (kt + 1) * 32;
      pa0 = *(const f32x4*)(ga);     pa1 = *(const f32x4*)(ga + 4);
      pa2 = *(const f32x4*)(ga + 8); pa3 = *(const f32x4*)(ga + 12);
      pw0 = *(const f32x4*)(gw);     pw1 = *(const f32x4*)(gw + 4);
      pw2 = *(const f32x4*)(gw + 8); pw3 = *(const f32x4*)(gw + 12);
    }

    // A fragments: ISA 16-bit A layout (lane half -> K {kh*8..+7} U {16+kh*8..+7})
    const bf16* bufA = sA[cur];
    const bf16* bufB = sB[cur];
    FragB16 af[2], bfr[4];
#pragma unroll
    for (int mi = 0; mi < 2; mi++) {
      const bf16* p = bufA + (wm + mi * 16 + l15) * 32 + khalf * 8;
      af[mi].h2[0] = *(const bf16x8*)p;
      af[mi].h2[1] = *(const bf16x8*)(p + 16);
    }
    // B fragments: same K order as A (lane half -> K {kh*8..+7} U {16+kh*8..+7}); W row = column n
#pragma unroll
    for (int ni = 0; ni < 4; ni++) {
      const bf16* p = bufB + (wn + ni * 16 + l15) * 32 + khalf * 8;
      bfr[ni].h2[0] = *(const bf16x8*)p;
      bfr[ni].h2[1] = *(const bf16x8*)(p + 16);
    }
#pragma unroll
    for (int mi = 0; mi < 2; mi++)
#pragma unroll
      for (int ni = 0; ni < 4; ni++)
        acc[mi][ni] = __builtin_amdgcn_wmma_f32_16x16x32_bf16(
            false, af[mi].v, false, bfr[ni].v, (short)0, acc[mi][ni], false, false);

    // Convert + store the prefetched tile into the other buffer.
    if (havenext) {
      bf16* dA = lA + nxt * (128 * 32);
      bf16* dW = lW + nxt * (128 * 32);
      *(bf16x8*)(dA)     = cvt8v(pa0, pa1);
      *(bf16x8*)(dA + 8) = cvt8v(pa2, pa3);
      *(bf16x8*)(dW)     = cvt8v(pw0, pw1);
      *(bf16x8*)(dW + 8) = cvt8v(pw2, pw3);
    }
    __syncthreads();
  }

  // Epilogue. C layout: row = r + 8*khalf (within 16-tile), col = l15.
#pragma unroll
  for (int mi = 0; mi < 2; mi++) {
#pragma unroll
    for (int ni = 0; ni < 4; ni++) {
      const int gc = colBase + wn + ni * 16 + l15;
      const float bv = bias[gc];
#pragma unroll
      for (int r = 0; r < 8; r++) {
        const int gr = rowBase + wm + mi * 16 + r + khalf * 8;
        float v = acc[mi][ni][r] + bv;
        if (EPI == EPI_QHEAD || EPI == EPI_KHEAD) {
          const int bt = gr / SEQ, sq = gr % SEQ;
          const int hh = gc >> 6, d = gc & 63;
          ((bf16*)out)[(((int64_t)bt * CHEADS + hh) * SEQ + sq) * CDQ + d] = (bf16)v;
        } else if (EPI == EPI_VT) { // store V transposed: vt[bth][d][a]
          const int bt = gr / SEQ, sq = gr % SEQ;
          const int hh = gc >> 6, d = gc & 63;
          ((bf16*)out)[(((int64_t)bt * CHEADS + hh) * CDQ + d) * SEQ + sq] = (bf16)v;
        } else if (EPI == EPI_BIAS_RES) {
          v += res[(int64_t)gr * N + gc];
          ((float*)out)[(int64_t)gr * N + gc] = v;
        } else { // EPI_BIAS_RELU
          ((float*)out)[(int64_t)gr * N + gc] = v > 0.f ? v : 0.f;
        }
      }
    }
  }
}

// Fused masked attention per (b,t,head): S = q k^T / 8, mask, softmax, ctx = P v.
// 8 waves; wave w owns query rows [16w, 16w+16).
__global__ __launch_bounds__(256)
void attn_kernel(const bf16* __restrict__ qbuf, const bf16* __restrict__ kbuf,
                 const bf16* __restrict__ vtbuf, const int* __restrict__ mask,
                 float* __restrict__ ctxbuf)
{
  __shared__ __align__(16) bf16 sP[8][16][64];

  const int bth  = blockIdx.x;
  const int bt   = bth >> 3;   // / HEADS
  const int hh   = bth & 7;    // % HEADS
  const int lane = threadIdx.x & 31;
  const int wave = threadIdx.x >> 5;
  const int l15  = lane & 15;
  const int khalf = lane >> 4;

  const bf16* qb = qbuf  + (int64_t)bth * (CL * CDQ);
  const bf16* kb = kbuf  + (int64_t)bth * (CA * CDQ);
  const bf16* vb = vtbuf + (int64_t)bth * (CDQ * CA);
  const int*  mb = mask  + (int64_t)bth * (CL * CA);

  // q A-fragments for this wave's 16-row strip (DQ=64 -> 2 K-steps of 32)
  FragB16 qf[2];
  {
    const bf16* qr = qb + (wave * 16 + l15) * CDQ;
#pragma unroll
    for (int ks = 0; ks < 2; ks++) {
      qf[ks].h2[0] = *(const bf16x8*)(qr + ks * 32 + khalf * 8);
      qf[ks].h2[1] = *(const bf16x8*)(qr + ks * 32 + 16 + khalf * 8);
    }
  }

  // Scores 16x64 (4 N-tiles of key columns)
  f32x8 s[4];
#pragma unroll
  for (int nt = 0; nt < 4; nt++) {
    f32x8 a = (f32x8){0.f, 0.f, 0.f, 0.f, 0.f, 0.f, 0.f, 0.f};
    const bf16* kr = kb + (nt * 16 + l15) * CDQ;  // B column n = key index a
#pragma unroll
    for (int ks = 0; ks < 2; ks++) {
      FragB16 kf;
      kf.h2[0] = *(const bf16x8*)(kr + ks * 32 + khalf * 8);
      kf.h2[1] = *(const bf16x8*)(kr + ks * 32 + 16 + khalf * 8);
      a = __builtin_amdgcn_wmma_f32_16x16x32_bf16(false, qf[ks].v, false, kf.v,
                                                  (short)0, a, false, false);
    }
    s[nt] = a;
  }

  // scale + mask
#pragma unroll
  for (int nt = 0; nt < 4; nt++)
#pragma unroll
    for (int r = 0; r < 8; r++) {
      const int row = wave * 16 + r + khalf * 8;
      const int col = nt * 16 + l15;
      const int m = mb[row * CA + col];
      const float v = s[nt][r] * 0.125f;
      s[nt][r] = (m == 0) ? -1e9f : v;
    }

  // softmax over 64 key cols: 4 tiles in-lane + 16-lane-half shfl reduction
  float p[4][8];
#pragma unroll
  for (int r = 0; r < 8; r++) {
    float mx = fmaxf(fmaxf(s[0][r], s[1][r]), fmaxf(s[2][r], s[3][r]));
#pragma unroll
    for (int o = 1; o < 16; o <<= 1) mx = fmaxf(mx, __shfl_xor(mx, o, 32));
    float sum = 0.f;
#pragma unroll
    for (int nt = 0; nt < 4; nt++) { float e = __expf(s[nt][r] - mx); p[nt][r] = e; sum += e; }
#pragma unroll
    for (int o = 1; o < 16; o <<= 1) sum += __shfl_xor(sum, o, 32);
    const float inv = 1.0f / sum;
#pragma unroll
    for (int nt = 0; nt < 4; nt++) p[nt][r] *= inv;
  }

  // reshape P (C layout) -> A-fragment layout via this wave's LDS strip
#pragma unroll
  for (int nt = 0; nt < 4; nt++)
#pragma unroll
    for (int r = 0; r < 8; r++)
      sP[wave][r + khalf * 8][nt * 16 + l15] = (bf16)p[nt][r];
  __syncthreads();

  FragB16 pf[2];
  {
    const bf16* pr = &sP[wave][l15][0];
#pragma unroll
    for (int ks = 0; ks < 2; ks++) {
      pf[ks].h2[0] = *(const bf16x8*)(pr + ks * 32 + khalf * 8);
      pf[ks].h2[1] = *(const bf16x8*)(pr + ks * 32 + 16 + khalf * 8);
    }
  }

  // ctx = P @ V  (B operand read from V^T rows: column n = feature d)
  float* cb = ctxbuf + ((int64_t)bt * CL) * CH + hh * CDQ;
#pragma unroll
  for (int nt = 0; nt < 4; nt++) {
    f32x8 a = (f32x8){0.f, 0.f, 0.f, 0.f, 0.f, 0.f, 0.f, 0.f};
    const bf16* vr = vb + (nt * 16 + l15) * CA;
#pragma unroll
    for (int ks = 0; ks < 2; ks++) {
      FragB16 vf;
      vf.h2[0] = *(const bf16x8*)(vr + ks * 32 + khalf * 8);
      vf.h2[1] = *(const bf16x8*)(vr + ks * 32 + 16 + khalf * 8);
      a = __builtin_amdgcn_wmma_f32_16x16x32_bf16(false, pf[ks].v, false, vf.v,
                                                  (short)0, a, false, false);
    }
#pragma unroll
    for (int r = 0; r < 8; r++) {
      const int row = wave * 16 + r + khalf * 8;
      const int col = nt * 16 + l15;
      cb[(int64_t)row * CH + col] = a[r];
    }
  }
}

// LayerNorm over last dim (512). One wave per row, 8 rows per block.
__global__ __launch_bounds__(256)
void layernorm_kernel(const float* __restrict__ x, const float* __restrict__ g,
                      const float* __restrict__ b, float* __restrict__ out)
{
  const int row  = blockIdx.x * 8 + (threadIdx.x >> 5);
  const int lane = threadIdx.x & 31;
  const float* xr = x + (int64_t)row * CH;

  f32x4 v[4];
  float sum = 0.f;
#pragma unroll
  for (int i = 0; i < 4; i++) {
    v[i] = *(const f32x4*)(xr + lane * 4 + i * 128);
    sum += v[i].x + v[i].y + v[i].z + v[i].w;
  }
#pragma unroll
  for (int o = 1; o < 32; o <<= 1) sum += __shfl_xor(sum, o, 32);
  const float mu = sum * (1.0f / 512.0f);

  float vs = 0.f;
#pragma unroll
  for (int i = 0; i < 4; i++)
#pragma unroll
    for (int j = 0; j < 4; j++) { const float d = v[i][j] - mu; vs += d * d; }
#pragma unroll
  for (int o = 1; o < 32; o <<= 1) vs += __shfl_xor(vs, o, 32);
  const float rstd = __frsqrt_rn(vs * (1.0f / 512.0f) + 1e-5f);

  float* orow = out + (int64_t)row * CH;
#pragma unroll
  for (int i = 0; i < 4; i++) {
    const int c = lane * 4 + i * 128;
    f32x4 gg = *(const f32x4*)(g + c);
    f32x4 bb = *(const f32x4*)(b + c);
    f32x4 o4;
#pragma unroll
    for (int j = 0; j < 4; j++) o4[j] = (v[i][j] - mu) * rstd * gg[j] + bb[j];
    *(f32x4*)(orow + c) = o4;
  }
}

extern "C" void kernel_launch(void* const* d_in, const int* in_sizes, int n_in,
                              void* d_out, int out_size, void* d_ws, size_t ws_size,
                              hipStream_t stream) {
  (void)in_sizes; (void)n_in; (void)out_size; (void)ws_size;
  const float* input_Q = (const float*)d_in[0];
  const float* input_K = (const float*)d_in[1];
  const float* input_V = (const float*)d_in[2];
  const int*   mask    = (const int*)d_in[3];
  const float* W_Q  = (const float*)d_in[4];
  const float* b_Q  = (const float*)d_in[5];
  const float* W_K  = (const float*)d_in[6];
  const float* b_K  = (const float*)d_in[7];
  const float* W_V  = (const float*)d_in[8];
  const float* b_V  = (const float*)d_in[9];
  const float* W_fc = (const float*)d_in[10];
  const float* b_fc = (const float*)d_in[11];
  const float* ln1_g = (const float*)d_in[12];
  const float* ln1_b = (const float*)d_in[13];
  const float* ln2_g = (const float*)d_in[14];
  const float* ln2_b = (const float*)d_in[15];
  const float* ff_w1 = (const float*)d_in[16];
  const float* ff_b1 = (const float*)d_in[17];
  const float* ff_w2 = (const float*)d_in[18];
  const float* ff_b2 = (const float*)d_in[19];

  // Workspace layout (bytes)
  char* ws = (char*)d_ws;
  bf16*  qbuf   = (bf16*)(ws);                        // [B*T*H][L][DQ]  bf16 : 209,715,200 B
  bf16*  kbuf   = (bf16*)(ws + 209715200ull);         // [B*T*H][A][DQ]  bf16 : 104,857,600 B
  bf16*  vtbuf  = (bf16*)(ws + 314572800ull);         // [B*T*H][DQ][A]  bf16 : 104,857,600 B
  float* ctxbuf = (float*)(ws + 419430400ull);        // [M][512] f32         : 419,430,400 B
  float* pre    = (float*)(ws + 838860800ull);        // pre-LN buffer f32    : 419,430,400 B
  float* x1     = (float*)(ws + 1258291200ull);       // post-LN1 f32         : 419,430,400 B
  float* h1     = (float*)(ws + 1677721600ull);       // FFN hidden f32       : 838,860,800 B

  dim3 blk(256);

  // Q/K/V projections with head-layout scatter epilogues (grid: cols fastest)
  gemm_wmma<EPI_QHEAD, CL><<<dim3(CH / 128, MQ / 128), blk, 0, stream>>>(
      input_Q, W_Q, b_Q, nullptr, qbuf, CH, CH);
  gemm_wmma<EPI_KHEAD, CA><<<dim3(CH / 128, MK / 128), blk, 0, stream>>>(
      input_K, W_K, b_K, nullptr, kbuf, CH, CH);
  gemm_wmma<EPI_VT, CA><<<dim3(CH / 128, MK / 128), blk, 0, stream>>>(
      input_V, W_V, b_V, nullptr, vtbuf, CH, CH);

  // Fused masked attention
  attn_kernel<<<CB * CT * CHEADS, blk, 0, stream>>>(qbuf, kbuf, vtbuf, mask, ctxbuf);

  // fc projection + residual, then LN1
  gemm_wmma<EPI_BIAS_RES, 1><<<dim3(CH / 128, MQ / 128), blk, 0, stream>>>(
      ctxbuf, W_fc, b_fc, input_Q, pre, CH, CH);
  layernorm_kernel<<<MQ / 8, blk, 0, stream>>>(pre, ln1_g, ln1_b, x1);

  // FFN: relu(x1 @ w1^T + b1) @ w2^T + b2 + x1, then LN2 -> d_out
  gemm_wmma<EPI_BIAS_RELU, 1><<<dim3(CDFF / 128, MQ / 128), blk, 0, stream>>>(
      x1, ff_w1, ff_b1, nullptr, h1, CDFF, CH);
  gemm_wmma<EPI_BIAS_RES, 1><<<dim3(CH / 128, MQ / 128), blk, 0, stream>>>(
      h1, ff_w2, ff_b2, x1, pre, CH, CDFF);
  layernorm_kernel<<<MQ / 8, blk, 0, stream>>>(pre, ln2_g, ln2_b, (float*)d_out);
}
